// EncoderME_75325136437391
// MI455X (gfx1250) — compile-verified, matrix operand order checked by reading, not run on hardware
//
#include <hip/hip_runtime.h>

// ---------------------------------------------------------------------------
// CDNA5 (gfx1250) WMMA types + async-to-LDS availability
// ---------------------------------------------------------------------------
typedef __attribute__((ext_vector_type(16))) __bf16 v16bf;
typedef __attribute__((ext_vector_type(8)))  float  v8f;

#if defined(__HIP_DEVICE_COMPILE__) && __has_builtin(__builtin_amdgcn_global_load_async_to_lds_b128)
#define HAVE_ASYNC_LDS 1
// builtin signature (from hipcc diagnostic): (v4i AS1*, v4i AS3*, Ii, Ii)
typedef int async_i4 __attribute__((vector_size(16)));
#endif

__device__ __forceinline__ void wait_async_all() {
#if defined(HAVE_ASYNC_LDS)
#if __has_builtin(__builtin_amdgcn_s_wait_asynccnt)
  __builtin_amdgcn_s_wait_asynccnt(0);
#else
  asm volatile("s_wait_asynccnt 0x0" ::: "memory");
#endif
#endif
}

__device__ __forceinline__ void copy16_g2lds(const __bf16* gp, __bf16* lp) {
#if defined(HAVE_ASYNC_LDS)
  // async DMA: GLOBAL_LOAD_ASYNC_TO_LDS_B128 (ASYNCcnt-tracked, no VGPR trip)
  __builtin_amdgcn_global_load_async_to_lds_b128(
      (__attribute__((address_space(1))) async_i4*)gp,
      (__attribute__((address_space(3))) async_i4*)lp, 0, 0);
#else
  *(uint4*)lp = *(const uint4*)gp;
#endif
}

// ---------------------------------------------------------------------------
// small elementwise kernels
// ---------------------------------------------------------------------------
__global__ void f32_to_bf16_k(const float* __restrict__ src,
                              __bf16* __restrict__ dst, int n) {
  int i = blockIdx.x * blockDim.x + threadIdx.x;
  if (i < n) dst[i] = (__bf16)src[i];
}

__global__ void zero_k(float* __restrict__ p, int n) {
  int i = blockIdx.x * blockDim.x + threadIdx.x;
  if (i < n) p[i] = 0.f;
}

// ---------------------------------------------------------------------------
// Implicit-GEMM convolution via v_wmma_f32_16x16x32_bf16.
//   M = Nn*Ho*Wo, K = KH*KW*Cin, N = Cout
// Block = 256 threads (8 waves). Wave tile = 64x16 (4 accumulators / 4 WMMA
// per K-chunk). Block tile = 512x16. Double-buffered LDS + async-to-LDS
// copies overlap staging of chunk t+1 with the WMMAs of chunk t.
// Fast path (Cin%32==0): a 32-wide K-chunk is one (kh,kw) cell -> each A row
// is 64 contiguous bytes in global -> 8 async b128 issues per lane.
// ---------------------------------------------------------------------------
__global__ __launch_bounds__(256) void conv_wmma_k(
    const __bf16* __restrict__ in, const __bf16* __restrict__ w,
    float* __restrict__ out,
    int Mtot, int Ktot, int Cout,
    int Hi, int Wi, int Cin, int lCin,
    int Ho, int lHo, int Wo, int lWo,
    int KW, int SH, int SW, int PH, int PW)
{
  const int tid  = threadIdx.x;
  const int wave = tid >> 5;
  const int lane = tid & 31;
  const int mBase = blockIdx.x * 512 + wave * 64;
  const int n0    = blockIdx.y * 16;

  __shared__ __align__(16) __bf16 a_lds[2][8 * 64 * 32];  // 2 x 32 KB
  __shared__ __align__(16) __bf16 b_lds[2][32 * 16];      // 2 x 1 KB

  const int HiWi   = Hi * Wi;
  const int maskWo = Wo - 1, maskHo = Ho - 1, maskCi = Cin - 1;
  const bool fast  = ((Cin & 31) == 0);

  // stage one 32-wide K-chunk (A: this wave's 64 rows; B: cooperative)
  auto stage = [&](int buf, int kkc) {
    __bf16* awb = a_lds[buf] + wave * 2048;
    if (fast) {
      const int cell = kkc >> lCin;            // uniform per chunk
      const int kh   = cell / KW;
      const int kw   = cell - kh * KW;
      const int dh   = kh - PH, dw = kw - PW;
      const int ci0  = kkc & maskCi;
#pragma unroll
      for (int p = 0; p < 8; ++p) {            // 64 rows x 4 x 16B pieces
        const int idx = lane + p * 32;
        const int j = idx >> 2, piece = idx & 3;
        const int gm = mBase + j;
        const int wo = gm & maskWo;
        const int t2 = gm >> lWo;
        const int ho = t2 & maskHo;
        const int nb = t2 >> lHo;
        const int hr = ho * SH + dh;
        const int wc = wo * SW + dw;
        __bf16* lp = awb + j * 32 + piece * 8;
        if (gm < Mtot &&
            (unsigned)hr < (unsigned)Hi && (unsigned)wc < (unsigned)Wi) {
          const __bf16* gp =
              in + (long)(nb * HiWi + hr * Wi + wc) * Cin + ci0 + piece * 8;
          copy16_g2lds(gp, lp);
        } else {
          uint4 z; z.x = z.y = z.z = z.w = 0u;
          *(uint4*)lp = z;                     // zero-fill padding rows
        }
      }
    } else {
      // generic gather (Cin in {1,16}: layers 1-3 only)
      const int gk   = kkc + lane;
      const bool kOK = gk < Ktot;
      const int ci   = gk & maskCi;
      const int cell = gk >> lCin;
      const int kh   = cell / KW;
      const int kw   = cell - kh * KW;
      const int dh   = kh - PH, dw = kw - PW;
      for (int j = 0; j < 64; ++j) {
        const int gm = mBase + j;
        const int wo = gm & maskWo;
        const int t2 = gm >> lWo;
        const int ho = t2 & maskHo;
        const int nb = t2 >> lHo;
        const int hr = ho * SH + dh;
        const int wc = wo * SW + dw;
        __bf16 v = (__bf16)0.0f;
        if (kOK && gm < Mtot &&
            (unsigned)hr < (unsigned)Hi && (unsigned)wc < (unsigned)Wi)
          v = in[(long)(nb * HiWi + hr * Wi + wc) * Cin + ci];
        awb[j * 32 + lane] = v;
      }
    }
    // B tile: 32 x 16, transposed into K-contiguous columns
    for (int i = tid; i < 512; i += 256) {
      const int k = i >> 4, n = i & 15;
      const int g = kkc + k;
      b_lds[buf][n * 32 + k] = (g < Ktot) ? w[(long)g * Cout + n0 + n]
                                          : (__bf16)0.0f;
    }
    // pull the chunk after next into L2 (global_prefetch_b8)
    if (lane == 0 && kkc + 32 < Ktot)
      __builtin_prefetch(&w[(long)(kkc + 32) * Cout + n0], 0, 1);
  };

  v8f acc[4] = {{}, {}, {}, {}};
  const int mr = lane & 15, hk = lane >> 4;
  union Frag { uint4 q[2]; v16bf v; } bfv, af;

  const int nch = (Ktot + 31) >> 5;
  stage(0, 0);
  for (int t = 0; t < nch; ++t) {
    const int cur = t & 1;
    wait_async_all();        // this wave's async A copies into buf[cur] done
    __syncthreads();         // everyone's B (ds) stores into buf[cur] done

    // fragment loads (16-bit WMMA VGPR layout: lanes 0-15 K {0..7},{16..23};
    // lanes 16-31 K {8..15},{24..31})
    const uint4* bp = (const uint4*)b_lds[cur];
    bfv.q[0] = bp[mr * 4 + hk];
    bfv.q[1] = bp[mr * 4 + hk + 2];
    const uint4* ap = (const uint4*)(a_lds[cur] + wave * 2048);
    uint4 aq[8];
#pragma unroll
    for (int s = 0; s < 4; ++s) {
      aq[2 * s]     = ap[(s * 16 + mr) * 4 + hk];
      aq[2 * s + 1] = ap[(s * 16 + mr) * 4 + hk + 2];
    }

    if (t + 1 < nch) stage(cur ^ 1, (t + 1) * 32);  // overlap with WMMA

#pragma unroll
    for (int s = 0; s < 4; ++s) {
      af.q[0] = aq[2 * s];
      af.q[1] = aq[2 * s + 1];
      acc[s] = __builtin_amdgcn_wmma_f32_16x16x32_bf16(
          false, af.v, false, bfv.v, (short)0, acc[s], false, false);
    }
  }

  // store: lane -> column N = lane&15; VGPR r -> row r (+8 for lanes 16-31)
  const int col = n0 + (lane & 15);
#pragma unroll
  for (int s = 0; s < 4; ++s) {
    const int rowBase = mBase + s * 16 + ((lane >> 4) << 3);
    for (int r = 0; r < 8; ++r) {
      const int row = rowBase + r;
      if (row < Mtot) out[(long)row * Cout + col] = acc[s][r];
    }
  }
}

// ---------------------------------------------------------------------------
// BN: contiguous grid-stride reads -> per-channel LDS bins (ds_add_f32)
// -> one global atomic per channel per block. C is always a power of two.
// ---------------------------------------------------------------------------
__global__ __launch_bounds__(256) void bn_partial_k(
    const float* __restrict__ h, float* __restrict__ sums,
    int total, int C, int maskC) {
  __shared__ float ls[2048];
  __shared__ float lq[2048];
  const int tid = threadIdx.x;
  for (int c = tid; c < C; c += 256) { ls[c] = 0.f; lq[c] = 0.f; }
  __syncthreads();
  const int stride = gridDim.x * 256;
  for (int i = blockIdx.x * 256 + tid; i < total; i += stride) {
    float v = h[i];
    int c = i & maskC;
    atomicAdd(&ls[c], v);
    atomicAdd(&lq[c], v * v);
  }
  __syncthreads();
  for (int c = tid; c < C; c += 256) {
    atomicAdd(&sums[c], ls[c]);
    atomicAdd(&sums[C + c], lq[c]);
  }
}

__global__ void bn_finalize_k(float* __restrict__ sb,
                              const float* __restrict__ gamma,
                              const float* __restrict__ beta, int M, int C) {
  int c = blockIdx.x * blockDim.x + threadIdx.x;
  if (c < C) {
    float mean = sb[c] / (float)M;
    float var  = sb[C + c] / (float)M - mean * mean;
    float sc   = gamma[c] * rsqrtf(var + 1e-5f);
    sb[c]     = sc;
    sb[C + c] = beta[c] - mean * sc;
  }
}

__global__ void bn_relu_bf16_k(const float* __restrict__ h,
                               const float* __restrict__ sb,
                               __bf16* __restrict__ o, int total, int maskC,
                               int C) {
  int i = blockIdx.x * blockDim.x + threadIdx.x;
  if (i < total) {
    int c = i & maskC;
    float v = h[i] * sb[c] + sb[C + c];
    o[i] = (__bf16)fmaxf(v, 0.f);
  }
}

__global__ void bias_relu_bf16_k(const float* __restrict__ h,
                                 const float* __restrict__ b,
                                 __bf16* __restrict__ o, int total, int maskC) {
  int i = blockIdx.x * blockDim.x + threadIdx.x;
  if (i < total) o[i] = (__bf16)fmaxf(h[i] + b[i & maskC], 0.f);
}

__global__ void bias_f32_k(const float* __restrict__ h,
                           const float* __restrict__ b,
                           float* __restrict__ o, int total, int maskC) {
  int i = blockIdx.x * blockDim.x + threadIdx.x;
  if (i < total) o[i] = h[i] + b[i & maskC];
}

// ---------------------------------------------------------------------------
// Host-side orchestration
// ---------------------------------------------------------------------------
static int ilog2i(int v) { int l = 0; while ((1 << l) < v) ++l; return l; }

extern "C" void kernel_launch(void* const* d_in, const int* in_sizes, int n_in,
                              void* d_out, int out_size, void* d_ws, size_t ws_size,
                              hipStream_t stream) {
  (void)in_sizes; (void)n_in; (void)out_size; (void)ws_size;

  // d_in order: x, conv_w[15], gamma[15], beta[15], lin_w[3], lin_b[3]
  const float* x = (const float*)d_in[0];
  const float* conv_w[15]; const float* gamma[15]; const float* beta[15];
  for (int i = 0; i < 15; ++i) {
    conv_w[i] = (const float*)d_in[1 + i];
    gamma[i]  = (const float*)d_in[16 + i];
    beta[i]   = (const float*)d_in[31 + i];
  }
  const float* lin_w[3]; const float* lin_b[3];
  for (int j = 0; j < 3; ++j) {
    lin_w[j] = (const float*)d_in[46 + j];
    lin_b[j] = (const float*)d_in[49 + j];
  }

  // workspace layout
  char* ws = (char*)d_ws;
  const size_t MB = (size_t)1 << 20;
  __bf16* act0  = (__bf16*)(ws + 0);         // 20 MB (bf16 activations, ping)
  __bf16* act1  = (__bf16*)(ws + 20 * MB);   // 20 MB (pong)
  float*  convf = (float*) (ws + 40 * MB);   // 36 MB (f32 conv out, pre-BN)
  __bf16* wb    = (__bf16*)(ws + 76 * MB);   // 42 MB (bf16 weights)
  float*  sb    = (float*) (ws + 118 * MB);  // BN sums/scale/bias scratch

  // layer specs (mirrors conv_specs())
  struct LSpec { int cin, cout, kh, kw, sh, sw, ph, pw; };
  const int CHv[8] = {16, 32, 64, 128, 256, 512, 1024, 2048};
  LSpec sp[15];
  {
    int idx = 0, inc = 1;
    for (int i = 0; i < 7; ++i) {
      sp[idx++] = {inc,    CHv[i], 2, 2, 2, 2, 0, 0};  // k2 s2 VALID
      sp[idx++] = {CHv[i], CHv[i], 3, 3, 1, 1, 1, 1};  // k3 s1 SAME
      inc = CHv[i];
    }
    sp[idx++] = {CHv[6], CHv[7], 2, 1, 2, 1, 0, 0};    // (2,1)/(2,1)
  }

  // bf16 weight pool offsets
  size_t woff[16]; woff[0] = 0;
  for (int i = 0; i < 15; ++i)
    woff[i + 1] = woff[i] + (size_t)sp[i].kh * sp[i].kw * sp[i].cin * sp[i].cout;
  const int lk[3] = {2048, 256, 128};
  const int ln[3] = {256, 128, 64};
  size_t loff[4]; loff[0] = woff[15];
  for (int j = 0; j < 3; ++j) loff[j + 1] = loff[j] + (size_t)lk[j] * ln[j];

  // convert all weights + input to bf16
  for (int i = 0; i < 15; ++i) {
    int n = (int)(woff[i + 1] - woff[i]);
    f32_to_bf16_k<<<(n + 255) / 256, 256, 0, stream>>>(conv_w[i], wb + woff[i], n);
  }
  for (int j = 0; j < 3; ++j) {
    int n = lk[j] * ln[j];
    f32_to_bf16_k<<<(n + 255) / 256, 256, 0, stream>>>(lin_w[j], wb + loff[j], n);
  }
  {
    int n = 32 * 256 * 256;
    f32_to_bf16_k<<<(n + 255) / 256, 256, 0, stream>>>(x, act0, n);
  }

  // conv trunk: conv -> BN stats -> BN+ReLU (bf16 out), ping-pong activations
  int Hi = 256, Wi = 256;
  __bf16* ain = act0; __bf16* aout = act1;
  for (int i = 0; i < 15; ++i) {
    LSpec s = sp[i];
    int Ho = (Hi + 2 * s.ph - s.kh) / s.sh + 1;
    int Wo = (Wi + 2 * s.pw - s.kw) / s.sw + 1;
    int M  = 32 * Ho * Wo;
    int Kt = s.kh * s.kw * s.cin;
    dim3 grid((M + 511) / 512, s.cout / 16);
    conv_wmma_k<<<grid, 256, 0, stream>>>(ain, wb + woff[i], convf,
        M, Kt, s.cout,
        Hi, Wi, s.cin, ilog2i(s.cin),
        Ho, ilog2i(Ho), Wo, ilog2i(Wo),
        s.kw, s.sh, s.sw, s.ph, s.pw);

    int C = s.cout, tot = M * C;
    zero_k<<<(2 * C + 255) / 256, 256, 0, stream>>>(sb, 2 * C);
    int nb = (tot + 255) / 256; if (nb > 256) nb = 256;
    bn_partial_k<<<nb, 256, 0, stream>>>(convf, sb, tot, C, C - 1);
    bn_finalize_k<<<(C + 255) / 256, 256, 0, stream>>>(sb, gamma[i], beta[i], M, C);
    bn_relu_bf16_k<<<(tot + 255) / 256, 256, 0, stream>>>(convf, sb, aout, tot, C - 1, C);

    Hi = Ho; Wi = Wo;
    __bf16* t = ain; ain = aout; aout = t;
  }

  // MLP head on [64, 2048]: same WMMA kernel as a 1x1 "conv" over M=64 pixels
  const int M = 64;
  for (int j = 0; j < 3; ++j) {
    dim3 grid((M + 511) / 512, ln[j] / 16);
    conv_wmma_k<<<grid, 256, 0, stream>>>(ain, wb + loff[j], convf,
        M, lk[j], ln[j],
        1, M, lk[j], ilog2i(lk[j]),
        1, 0, M, ilog2i(M),
        1, 1, 1, 0, 0);
    int tot = M * ln[j];
    if (j < 2) {
      bias_relu_bf16_k<<<(tot + 255) / 256, 256, 0, stream>>>(convf, lin_b[j], aout, tot, ln[j] - 1);
      __bf16* t = ain; ain = aout; aout = t;
    } else {
      bias_f32_k<<<(tot + 255) / 256, 256, 0, stream>>>(convf, lin_b[j], (float*)d_out, tot, ln[j] - 1);
    }
  }
}
